// TtMoeLayer_29875792511046
// MI455X (gfx1250) — matrix operand order repaired and checked by reading, not compiled, not run on hardware
//
#include <hip/hip_runtime.h>

typedef __bf16 bf16;
typedef __attribute__((ext_vector_type(4)))  float v4f;
typedef __attribute__((ext_vector_type(8)))  float v8f;
typedef __attribute__((ext_vector_type(4)))  bf16  v4bf;
typedef __attribute__((ext_vector_type(8)))  bf16  v8bf;
typedef __attribute__((ext_vector_type(16))) bf16  v16bf;
typedef int v4i __attribute__((vector_size(16)));   // pointee type of the async-LDS builtin

#define BDIM   8192
#define HDIM   1024
#define EDIM   8
#define MT     256      // M tile (tokens)
#define NT     128      // N tile (output features)
#define KT     64       // K tile
#define LPAD   8        // LDS row padding -> stride 72 elems = 144B (16B aligned)
#define KTILES (HDIM / KT)

union FragU { v16bf v; v8bf h[2]; };

// ---- CDNA5 async global->LDS copy (bypasses VGPR staging, tracked by ASYNCcnt)
__device__ __forceinline__ void cp_async16(void* lds, const void* gsrc) {
#if __has_builtin(__builtin_amdgcn_global_load_async_to_lds_b128)
    __builtin_amdgcn_global_load_async_to_lds_b128(
        (__attribute__((address_space(1))) v4i*)(void*)gsrc,
        (__attribute__((address_space(3))) v4i*)lds,
        0, 0);
#else
    *(v8bf*)lds = *(const v8bf*)gsrc;
#endif
}

__device__ __forceinline__ void wait_async0() {
#if __has_builtin(__builtin_amdgcn_s_wait_asynccnt)
    __builtin_amdgcn_s_wait_asynccnt(0);
#elif defined(__gfx1250__)
    asm volatile("s_wait_asynccnt 0x0" ::: "memory");
#endif
}

// ---------------------------------------------------------------- convert x
__global__ void convert_x_kernel(const float* __restrict__ x, bf16* __restrict__ xb) {
    int idx = blockIdx.x * blockDim.x + threadIdx.x;          // one thread = 4 elems
    if (idx >= (BDIM * HDIM) / 4) return;
    v4f  f = ((const v4f*)x)[idx];
    v4bf b;
    b.x = (bf16)f.x; b.y = (bf16)f.y; b.z = (bf16)f.z; b.w = (bf16)f.w;
    ((v4bf*)xb)[idx] = b;
}

// ------------------------------------------- transpose + convert expert_w
// W[e][h][o] f32  ->  Wt[e][o][h] bf16  (so GEMM B-tiles are K-contiguous)
__global__ void transpose_w_kernel(const float* __restrict__ W, bf16* __restrict__ Wt) {
    __shared__ bf16 tile[32][33];
    const int e  = blockIdx.z;
    const int i0 = blockIdx.y * 32;   // h
    const int j0 = blockIdx.x * 32;   // o
    const float* Wp  = W  + (size_t)e * HDIM * HDIM;
    bf16*        Wtp = Wt + (size_t)e * HDIM * HDIM;
    const int tx = threadIdx.x;       // 32
    const int ty = threadIdx.y;       // 8
#pragma unroll
    for (int k = 0; k < 4; ++k)
        tile[ty + 8 * k][tx] = (bf16)Wp[(size_t)(i0 + ty + 8 * k) * HDIM + (j0 + tx)];
    __syncthreads();
#pragma unroll
    for (int k = 0; k < 4; ++k)
        Wtp[(size_t)(j0 + ty + 8 * k) * HDIM + (i0 + tx)] = tile[tx][ty + 8 * k];
}

// ---------------------------------------------------------------- gate + route
__global__ void gate_route_kernel(const float* __restrict__ x,
                                  const float* __restrict__ gate_w,
                                  int* __restrict__ counts,
                                  int* __restrict__ tok_ids,
                                  float* __restrict__ gate_vals) {
    const int wid  = threadIdx.x >> 5;
    const int lane = threadIdx.x & 31;
    const int b    = blockIdx.x * 8 + wid;     // one wave per token
    if (b >= BDIM) return;

    float acc[EDIM];
#pragma unroll
    for (int e = 0; e < EDIM; ++e) acc[e] = 0.f;

    const float* xr = x + (size_t)b * HDIM;
    for (int h = lane; h < HDIM; h += 32) {
        float xv = xr[h];
        const float* gr = gate_w + (size_t)h * EDIM;
#pragma unroll
        for (int e = 0; e < EDIM; ++e) acc[e] += xv * gr[e];
    }
#pragma unroll
    for (int off = 16; off > 0; off >>= 1) {
#pragma unroll
        for (int e = 0; e < EDIM; ++e) acc[e] += __shfl_xor(acc[e], off, 32);
    }
    if (lane == 0) {
        int e1 = 0; float l1 = acc[0];
#pragma unroll
        for (int e = 1; e < EDIM; ++e) if (acc[e] > l1) { l1 = acc[e]; e1 = e; }
        int e2 = -1; float l2 = -3.0e38f;
#pragma unroll
        for (int e = 0; e < EDIM; ++e) if (e != e1 && acc[e] > l2) { l2 = acc[e]; e2 = e; }
        float t  = __expf(l2 - l1);
        float w1 = 1.f / (1.f + t);
        float w2 = 1.f - w1;
        int s1 = atomicAdd(&counts[e1], 1);
        tok_ids[e1 * BDIM + s1]   = b;
        gate_vals[e1 * BDIM + s1] = w1;
        int s2 = atomicAdd(&counts[e2], 1);
        tok_ids[e2 * BDIM + s2]   = b;
        gate_vals[e2 * BDIM + s2] = w2;
    }
}

// ---------------------------------------------------------------- expert GEMM
// C[tok, n] += gate * sum_k xb[tok,k] * Wt[e][n][k]   (bf16 WMMA, f32 acc)
// 256x128 block tile, 8 waves of 64x64, double-buffered async LDS pipeline.
__global__ __launch_bounds__(256)
void expert_gemm_kernel(const bf16* __restrict__ xb,
                        const bf16* __restrict__ wt,
                        const int* __restrict__ counts,
                        const int* __restrict__ tok_ids,
                        const float* __restrict__ gate_vals,
                        float* __restrict__ out) {
    const int e   = blockIdx.z;
    const int n0  = blockIdx.y * NT;
    const int m0  = blockIdx.x * MT;
    const int cnt = counts[e];
    if (m0 >= cnt) return;                      // uniform: whole block exits

    __shared__ __align__(16) bf16 As[2][MT][KT + LPAD];
    __shared__ __align__(16) bf16 Bs[2][NT][KT + LPAD];
    __shared__ int   toks[MT];
    __shared__ float gws[MT];

    const int t    = threadIdx.x;
    const int wid  = t >> 5;
    const int lane = t & 31;
    const int half = lane >> 4;
    const int l15  = lane & 15;
    const int wm   = (wid >> 1) * 64;           // wave M offset: 0,64,128,192
    const int wn   = (wid & 1) * 64;            // wave N offset: 0,64

    {   // routing metadata for this tile (MT == blockDim.x)
        int m = m0 + t;
        bool v = m < cnt;
        toks[t] = v ? tok_ids[e * BDIM + m] : 0;   // invalid rows read token 0 (finite junk)
        gws[t]  = v ? gate_vals[e * BDIM + m] : 0.f;
    }
    __syncthreads();

    // cache this thread's copy-row tokens in registers
    const int lrow = t >> 3;                    // 0..31
    const int lcol = (t & 7) * 8;               // 0..56, 16B granules
    int tokA[MT / 32];
#pragma unroll
    for (int r = 0; r < MT / 32; ++r) tokA[r] = toks[lrow + 32 * r];

    const bf16* wte = wt + (size_t)e * HDIM * HDIM;

    auto issue_tile = [&](int kt, int buf) {
        const int k0 = kt * KT;
#pragma unroll
        for (int r = 0; r < MT / 32; ++r) {
            int row = lrow + 32 * r;
            cp_async16(&As[buf][row][lcol],
                       xb + (size_t)tokA[r] * HDIM + k0 + lcol);
        }
#pragma unroll
        for (int r = 0; r < NT / 32; ++r) {
            int row = lrow + 32 * r;
            cp_async16(&Bs[buf][row][lcol],
                       wte + (size_t)(n0 + row) * HDIM + k0 + lcol);
        }
    };

    v8f acc[4][4];
#pragma unroll
    for (int i = 0; i < 4; ++i)
#pragma unroll
        for (int j = 0; j < 4; ++j) { v8f z = {}; acc[i][j] = z; }

    issue_tile(0, 0);

    for (int kt = 0; kt < KTILES; ++kt) {
        const int cur = kt & 1;
        wait_async0();                  // my copies for buf[cur] have landed in LDS
        __syncthreads();                // everyone's copies landed; prev compute done
        if (kt + 1 < KTILES) issue_tile(kt + 1, cur ^ 1);   // overlap with compute

#pragma unroll
        for (int ks = 0; ks < KT; ks += 32) {
            FragU af[4];
#pragma unroll
            for (int i = 0; i < 4; ++i) {
                const bf16* rp = &As[cur][wm + 16 * i + l15][0];
                af[i].h[0] = *(const v8bf*)(rp + ks + 16 * half);      // K 0-7 / 16-23
                af[i].h[1] = *(const v8bf*)(rp + ks + 8 + 16 * half);  // K 8-15 / 24-31
            }
            FragU bfr[4];
#pragma unroll
            for (int j = 0; j < 4; ++j) {
                const v8bf* p =
                    (const v8bf*)(&Bs[cur][wn + 16 * j + l15][0] + ks + 16 * half);
                bfr[j].h[0] = p[0];                                    // K 0-15 / 16-31
                bfr[j].h[1] = p[1];
            }
#pragma unroll
            for (int i = 0; i < 4; ++i)
#pragma unroll
                for (int j = 0; j < 4; ++j)
                    acc[i][j] = __builtin_amdgcn_wmma_f32_16x16x32_bf16(
                        false, af[i].v, false, bfr[j].v,
                        (short)0, acc[i][j], false, false);
        }
        __syncthreads();                // done reading buf[cur] before it is refilled
    }

    // ---- weighted scatter: out[tok, n] += gate * acc
#pragma unroll
    for (int i = 0; i < 4; ++i) {
#pragma unroll
        for (int r = 0; r < 8; ++r) {
            int ml = wm + 16 * i + r + 8 * half;   // C layout: vgpr r, half selects M+8
            if (m0 + ml < cnt) {
                int   tok = toks[ml];
                float g   = gws[ml];
                float* orow = out + (size_t)tok * HDIM + n0;
#pragma unroll
                for (int j = 0; j < 4; ++j)
                    atomicAdd(&orow[wn + 16 * j + l15], g * acc[i][j][r]);
            }
        }
    }
}

// ---------------------------------------------------------------- launcher
extern "C" void kernel_launch(void* const* d_in, const int* in_sizes, int n_in,
                              void* d_out, int out_size, void* d_ws, size_t ws_size,
                              hipStream_t stream) {
    const float* x   = (const float*)d_in[0];   // [B,H]
    const float* gw  = (const float*)d_in[1];   // [H,E]
    const float* ew  = (const float*)d_in[2];   // [E,H,H]
    float* out = (float*)d_out;                 // [B,H]

    char* ws = (char*)d_ws;
    bf16*  xb        = (bf16*)ws;                                   // 16 MB
    bf16*  wt        = (bf16*)(ws + ((size_t)16 << 20));            // 16 MB
    int*   counts    = (int*)(ws + ((size_t)32 << 20));             // 32 B (padded to 1 KB)
    int*   tok_ids   = (int*)(ws + ((size_t)32 << 20) + 1024);      // 256 KB
    float* gate_vals = (float*)(ws + ((size_t)32 << 20) + 1024 + (size_t)EDIM * BDIM * 4);

    (void)hipMemsetAsync(counts, 0, EDIM * sizeof(int), stream);
    (void)hipMemsetAsync(out, 0, (size_t)BDIM * HDIM * sizeof(float), stream);

    // 1) x -> bf16
    {
        int n = (BDIM * HDIM) / 4;
        convert_x_kernel<<<(n + 255) / 256, 256, 0, stream>>>(x, xb);
    }
    // 2) expert_w -> bf16, transposed to [E][o][h]
    {
        dim3 g(HDIM / 32, HDIM / 32, EDIM), b(32, 8);
        transpose_w_kernel<<<g, b, 0, stream>>>(ew, wt);
    }
    // 3) gate + routing lists
    gate_route_kernel<<<BDIM / 8, 256, 0, stream>>>(x, gw, counts, tok_ids, gate_vals);
    // 4) grouped expert GEMMs with weighted scatter
    {
        dim3 g(BDIM / MT, HDIM / NT, EDIM);
        expert_gemm_kernel<<<g, 256, 0, stream>>>(xb, wt, counts, tok_ids, gate_vals, out);
    }
}
